// WhisperBlock_17927193493851
// MI455X (gfx1250) — matrix operand order repaired and checked by reading, not compiled, not run on hardware
//
#include <hip/hip_runtime.h>
#include <hip/hip_bf16.h>
#include <math.h>

// ---------------------------------------------------------------- types
typedef __bf16 bf16;
typedef __attribute__((ext_vector_type(16))) __bf16 v16bf;
typedef __attribute__((ext_vector_type(8)))  float  v8f;
typedef unsigned int uint32x4 __attribute__((ext_vector_type(4)));
typedef int          int32x8  __attribute__((ext_vector_type(8)));
typedef int          int32x4  __attribute__((ext_vector_type(4)));

// ---------------------------------------------------------------- problem dims
#define B_   8
#define S_   1500
#define SP   1504          // S padded to multiple of 16
#define D_   1024
#define H_   16
#define HD_  64
#define DFF_ 4096
#define BS_  (B_ * S_)     // 12000 = 96 * 125

// GEMM tiling
#define BM 96
#define BN 128
#define BK 32

static_assert(BS_ % BM == 0, "M tiling");
static_assert(D_ % BN == 0 && DFF_ % BN == 0, "N tiling");
static_assert(D_ % BK == 0 && DFF_ % BK == 0, "K tiling");

#if __has_builtin(__builtin_amdgcn_tensor_load_to_lds)
#define USE_TDM 1
#else
#define USE_TDM 0
#endif

__device__ __forceinline__ void wait_tensorcnt0() {
#if __has_builtin(__builtin_amdgcn_s_wait_tensorcnt)
    __builtin_amdgcn_s_wait_tensorcnt(0);
#else
    asm volatile("s_wait_tensorcnt 0x0" ::: "memory");
#endif
}

// ---------------------------------------------------------------- helpers
__device__ __forceinline__ v8f wmma_bf16(v16bf a, v16bf b, v8f c) {
    // v_wmma_f32_16x16x32_bf16: D = A(16x32) * B(32x16) + C(16x16 f32)
    return __builtin_amdgcn_wmma_f32_16x16x32_bf16(
        /*neg_a=*/false, a, /*neg_b=*/false, b,
        /*c_mod=*/(short)0, c, /*reuse_a=*/false, /*reuse_b=*/false);
}

// Build a 16-element bf16 fragment from two 16-byte chunks.
__device__ __forceinline__ v16bf ld_frag2(const bf16* p0, const bf16* p1) {
    union { uint4 q[2]; v16bf v; } u;
    u.q[0] = *(const uint4*)p0;
    u.q[1] = *(const uint4*)p1;
    return u.v;
}

__device__ __forceinline__ v8f zero_v8f() {
    v8f z;
#pragma unroll
    for (int i = 0; i < 8; ++i) z[i] = 0.0f;
    return z;
}

#if USE_TDM
// Tensor Data Mover: DMA a fully in-bounds 2D bf16 tile (tile_d1 rows of
// tile_d0 elements, row stride = row_stride elements) from global into LDS
// at byte offset lds_off, packed row-major [tile_d1][tile_d0].
// D# bitfields per CDNA5 ISA.
__device__ __forceinline__ void tdm_load_tile_2d(const bf16* gptr, unsigned lds_off,
                                                 int tile_d0, int tile_d1,
                                                 int row_stride) {
    const unsigned long long ga = (unsigned long long)(size_t)gptr;
    uint32x4 g0;
    g0[0] = 1u;                                       // count=1, user mode
    g0[1] = lds_off;                                  // lds_addr (bytes)
    g0[2] = (unsigned)(ga & 0xffffffffu);             // global_addr[31:0]
    g0[3] = (unsigned)((ga >> 32) & 0x01ffffffu)      // global_addr[56:32]
          | 0x80000000u;                              // type=2 ("image")
    int32x8 g1;
    g1[0] = 1 << 16;                                  // data_size=1 -> 2 bytes
    g1[1] = (int)((unsigned)(tile_d0 & 0xffff) << 16);          // tensor_dim0 lo
    g1[2] = (int)(((unsigned)tile_d0 >> 16) |
                  ((unsigned)(tile_d1 & 0xffff) << 16));        // td0 hi | td1 lo
    g1[3] = (int)(((unsigned)tile_d1 >> 16) |
                  ((unsigned)(tile_d0 & 0xffff) << 16));        // td1 hi | tile_dim0
    g1[4] = (int)((unsigned)tile_d1 & 0xffff);                  // tile_dim1, tile_dim2=0
    g1[5] = row_stride;                                         // tensor_dim0_stride[31:0]
    g1[6] = 0;                                                  // stride hi, dim1_stride lo
    g1[7] = 0;
    const int32x4 z4 = {0, 0, 0, 0};                  // groups 2/3: 2-D tensor
    int32x8 z8;
#pragma unroll
    for (int i = 0; i < 8; ++i) z8[i] = 0;
    __builtin_amdgcn_tensor_load_to_lds(g0, g1, z4, z4, z8, 0);
}
#endif

// ---------------------------------------------------------------- f32 -> bf16
__global__ __launch_bounds__(256) void k_cvt_bf16(const float* __restrict__ src,
                                                  bf16* __restrict__ dst, int n) {
    int i = blockIdx.x * 256 + threadIdx.x;
    if (i < n) dst[i] = (bf16)src[i];
}

// ---------------------------------------------------------------- LayerNorm (per token)
__global__ __launch_bounds__(256) void k_layernorm(const float* __restrict__ x,
                                                   const float* __restrict__ g,
                                                   const float* __restrict__ b,
                                                   bf16* __restrict__ out) {
    __shared__ float r1[8], r2[8];
    const int t   = blockIdx.x;
    const int tid = threadIdx.x;
    const float4 v = ((const float4*)(x + (size_t)t * D_))[tid];
    float s = v.x + v.y + v.z + v.w;
    float q = v.x * v.x + v.y * v.y + v.z * v.z + v.w * v.w;
#pragma unroll
    for (int m = 16; m >= 1; m >>= 1) {
        s += __shfl_xor(s, m, 32);
        q += __shfl_xor(q, m, 32);
    }
    if ((tid & 31) == 0) { r1[tid >> 5] = s; r2[tid >> 5] = q; }
    __syncthreads();
    float S1 = 0.f, S2 = 0.f;
#pragma unroll
    for (int i = 0; i < 8; ++i) { S1 += r1[i]; S2 += r2[i]; }
    const float mean = S1 * (1.0f / D_);
    const float var  = S2 * (1.0f / D_) - mean * mean;
    const float rstd = rsqrtf(var + 1e-5f);
    const float4 gg = ((const float4*)g)[tid];
    const float4 bb = ((const float4*)b)[tid];
    bf16* o = out + (size_t)t * D_ + tid * 4;
    o[0] = (bf16)((v.x - mean) * rstd * gg.x + bb.x);
    o[1] = (bf16)((v.y - mean) * rstd * gg.y + bb.y);
    o[2] = (bf16)((v.z - mean) * rstd * gg.z + bb.z);
    o[3] = (bf16)((v.w - mean) * rstd * gg.w + bb.w);
}

// ---------------------------------------------------------------- WMMA GEMM
// C[M,N] = A[M,K] @ W[N,K]^T  (torch Linear). Epilogue by MODE:
//   0: scale, write bf16 in head layout [B,H,SP,64]      (Q / K)
//   1: write bf16 transposed head layout [B,H,64,SP]     (V)
//   2: += res (f32), write f32                           (attn out proj)
//   3: += bias, exact GELU, write bf16                   (FFN1)
//   4: += bias + res (f32), write f32                    (FFN2 -> d_out)
template <int MODE>
__global__ __launch_bounds__(256) void k_gemm(const bf16* __restrict__ A,
                                              const bf16* __restrict__ W,
                                              const float* __restrict__ bias,
                                              const float* __restrict__ res,
                                              void* __restrict__ outp,
                                              int M, int N, int K, float scale) {
    __shared__ bf16 As[BM * BK];
    __shared__ bf16 Bs[BN * BK];

    const int tid  = threadIdx.x;
    const int wid  = tid >> 5;
    const int lane = tid & 31;
    const int ln   = lane & 15;
    const int lh   = lane >> 4;

    const int m0 = blockIdx.y * BM;
    const int n0 = blockIdx.x * BN;
    const int wm = (wid >> 2) * 48;   // wave row offset within block tile
    const int wn = (wid & 3) * 32;    // wave col offset within block tile

    v8f acc[3][2];
#pragma unroll
    for (int i = 0; i < 3; ++i)
#pragma unroll
        for (int j = 0; j < 2; ++j) acc[i][j] = zero_v8f();

    for (int k0 = 0; k0 < K; k0 += BK) {
#if USE_TDM
        // Tensor Data Mover stages both tiles; wave 0 issues, all waves sync.
        if (wid == 0) {
            tdm_load_tile_2d(&A[(size_t)m0 * K + k0],
                             (unsigned)(size_t)(void*)As, BK, BM, K);
            tdm_load_tile_2d(&W[(size_t)n0 * K + k0],
                             (unsigned)(size_t)(void*)Bs, BK, BN, K);
            wait_tensorcnt0();
        }
        __syncthreads();
#else
        // cooperative stage: A tile 96x32 (384 chunks), B tile 128x32 (512 chunks)
#pragma unroll 2
        for (int c = tid; c < (BM * BK) / 8; c += 256) {
            const int row = c >> 2, kc = (c & 3) * 8;
            *(uint4*)&As[row * BK + kc] =
                *(const uint4*)&A[(size_t)(m0 + row) * K + k0 + kc];
        }
#pragma unroll 2
        for (int c = tid; c < (BN * BK) / 8; c += 256) {
            const int row = c >> 2, kc = (c & 3) * 8;
            *(uint4*)&Bs[row * BK + kc] =
                *(const uint4*)&W[(size_t)(n0 + row) * K + k0 + kc];
        }
        __syncthreads();
#endif

        v16bf af[3], bfr[2];
#pragma unroll
        for (int i = 0; i < 3; ++i) {
            const bf16* p = &As[(wm + i * 16 + ln) * BK + lh * 8];
            af[i] = ld_frag2(p, p + 16);           // A-layout: K chunks at k, k+16
        }
#pragma unroll
        for (int j = 0; j < 2; ++j) {
            const bf16* p = &Bs[(wn + j * 16 + ln) * BK + lh * 8];
            bfr[j] = ld_frag2(p, p + 16);          // B-layout: K chunks at k, k+16
        }
#pragma unroll
        for (int i = 0; i < 3; ++i)
#pragma unroll
            for (int j = 0; j < 2; ++j) acc[i][j] = wmma_bf16(af[i], bfr[j], acc[i][j]);
        __syncthreads();
    }

    // epilogue: lane holds column n = ln, rows mbase..mbase+7 (mbase = lh*8)
#pragma unroll
    for (int i = 0; i < 3; ++i)
#pragma unroll
        for (int j = 0; j < 2; ++j) {
            const int n  = n0 + wn + j * 16 + ln;
            const int mb = m0 + wm + i * 16 + lh * 8;
#pragma unroll
            for (int r = 0; r < 8; ++r) {
                const int m = mb + r;
                const float c = acc[i][j][r];
                if (MODE == 0) {
                    const int b = m / S_, s = m % S_;
                    const int h = n >> 6, hd = n & 63;
                    ((bf16*)outp)[((size_t)(b * H_ + h) * SP + s) * HD_ + hd] =
                        (bf16)(c * scale);
                } else if (MODE == 1) {
                    const int b = m / S_, s = m % S_;
                    const int h = n >> 6, hd = n & 63;
                    ((bf16*)outp)[((size_t)(b * H_ + h) * HD_ + hd) * SP + s] = (bf16)c;
                } else if (MODE == 2) {
                    ((float*)outp)[(size_t)m * N + n] = res[(size_t)m * N + n] + c;
                } else if (MODE == 3) {
                    const float t = c + bias[n];
                    const float gel = 0.5f * t * (1.0f + erff(t * 0.70710678f));
                    ((bf16*)outp)[(size_t)m * N + n] = (bf16)gel;
                } else {
                    ((float*)outp)[(size_t)m * N + n] =
                        res[(size_t)m * N + n] + c + bias[n];
                }
            }
        }
}

// ---------------------------------------------------------------- flash attention
// q,k: [B,H,SP,64] bf16 (q pre-scaled); vt: [B,H,64,SP] bf16; ctx: [B,S,D] bf16
// One wave per 16-row Q tile; online softmax over K in chunks of 32 columns.
#define NQT (SP / 16)   // 94
#define NQG 12          // ceil(94/8) groups of 8 waves

__global__ __launch_bounds__(256) void k_attn(const bf16* __restrict__ q,
                                              const bf16* __restrict__ k,
                                              const bf16* __restrict__ vt,
                                              bf16* __restrict__ ctx) {
    __shared__ bf16 pbuf_all[8][16 * 32];
    const int wid  = threadIdx.x >> 5;
    const int lane = threadIdx.x & 31;
    const int ln   = lane & 15;
    const int lh   = lane >> 4;

    const int bh = blockIdx.x / NQG;
    const int qt = (blockIdx.x % NQG) * 8 + wid;
    if (qt >= NQT) return;                    // no block-wide barriers below

    bf16* pbuf = pbuf_all[wid];
    const bf16* qb = q  + (size_t)bh * SP * HD_;
    const bf16* kb = k  + (size_t)bh * SP * HD_;
    const bf16* vb = vt + (size_t)bh * HD_ * SP;

    // Q fragments (16x64 = two 16x32 A-frags), loaded once
    const bf16* qp = qb + (size_t)(qt * 16 + ln) * HD_ + lh * 8;
    const v16bf qf0 = ld_frag2(qp,      qp + 16);
    const v16bf qf1 = ld_frag2(qp + 32, qp + 48);

    float mrun[8], lrun[8];
    v8f accd[4];
#pragma unroll
    for (int r = 0; r < 8; ++r) { mrun[r] = -3.0e38f; lrun[r] = 0.0f; }
#pragma unroll
    for (int t = 0; t < 4; ++t) accd[t] = zero_v8f();

    for (int j0 = 0; j0 < SP; j0 += 32) {
        // scores: two 16x16 C-tiles (columns j0..j0+15, j0+16..j0+31)
        v8f sc[2];
#pragma unroll
        for (int t = 0; t < 2; ++t) {
            sc[t] = zero_v8f();
            const bf16* kp = kb + (size_t)(j0 + t * 16 + ln) * HD_ + lh * 8;
            const v16bf kf0 = ld_frag2(kp,      kp + 16);
            const v16bf kf1 = ld_frag2(kp + 32, kp + 48);
            sc[t] = wmma_bf16(qf0, kf0, sc[t]);
            sc[t] = wmma_bf16(qf1, kf1, sc[t]);
        }
        // mask padded key columns
#pragma unroll
        for (int t = 0; t < 2; ++t)
            if (j0 + t * 16 + ln >= S_)
#pragma unroll
                for (int r = 0; r < 8; ++r) sc[t][r] = -3.0e38f;

        // online softmax: row max / rescale / row sum (rows live across 16 lanes)
        float alpha[8];
#pragma unroll
        for (int r = 0; r < 8; ++r) {
            float v = fmaxf(sc[0][r], sc[1][r]);
#pragma unroll
            for (int msk = 1; msk < 16; msk <<= 1) v = fmaxf(v, __shfl_xor(v, msk, 32));
            const float mnew = fmaxf(mrun[r], v);
            alpha[r] = __expf(mrun[r] - mnew);
            mrun[r]  = mnew;
            const float p0 = __expf(sc[0][r] - mnew);
            const float p1 = __expf(sc[1][r] - mnew);
            sc[0][r] = p0; sc[1][r] = p1;
            float s2 = p0 + p1;
#pragma unroll
            for (int msk = 1; msk < 16; msk <<= 1) s2 += __shfl_xor(s2, msk, 32);
            lrun[r] = lrun[r] * alpha[r] + s2;
        }
#pragma unroll
        for (int t = 0; t < 4; ++t)
#pragma unroll
            for (int r = 0; r < 8; ++r) accd[t][r] *= alpha[r];

        // C-layout -> A-layout for P via wave-private LDS
#pragma unroll
        for (int t = 0; t < 2; ++t)
#pragma unroll
            for (int r = 0; r < 8; ++r)
                pbuf[(lh * 8 + r) * 32 + t * 16 + ln] = (bf16)sc[t][r];
        __builtin_amdgcn_wave_barrier();
        const bf16* pp = &pbuf[ln * 32 + lh * 8];
        const v16bf pf = ld_frag2(pp, pp + 16);

        // accd += P(16x32) @ V(32x64), V fragments from transposed layout
#pragma unroll
        for (int t = 0; t < 4; ++t) {
            const bf16* vp = vb + (size_t)(t * 16 + ln) * SP + j0 + lh * 8;
            const v16bf vf = ld_frag2(vp, vp + 16);
            accd[t] = wmma_bf16(pf, vf, accd[t]);
        }
    }

    // normalize and store ctx [B,S,D]
    const int b = bh >> 4, h = bh & 15;
#pragma unroll
    for (int t = 0; t < 4; ++t) {
        const int d = h * HD_ + t * 16 + ln;
#pragma unroll
        for (int r = 0; r < 8; ++r) {
            const int s = qt * 16 + lh * 8 + r;
            if (s < S_)
                ctx[((size_t)(b * S_ + s)) * D_ + d] = (bf16)(accd[t][r] / lrun[r]);
        }
    }
}

// ---------------------------------------------------------------- launcher
extern "C" void kernel_launch(void* const* d_in, const int* in_sizes, int n_in,
                              void* d_out, int out_size, void* d_ws, size_t ws_size,
                              hipStream_t stream) {
    const float* hs    = (const float*)d_in[0];
    const float* Wq    = (const float*)d_in[1];
    const float* Wk    = (const float*)d_in[2];
    const float* Wv    = (const float*)d_in[3];
    const float* Wo    = (const float*)d_in[4];
    const float* W1    = (const float*)d_in[5];
    const float* b1    = (const float*)d_in[6];
    const float* W2    = (const float*)d_in[7];
    const float* b2    = (const float*)d_in[8];
    const float* ln1_g = (const float*)d_in[9];
    const float* ln1_b = (const float*)d_in[10];
    const float* ln2_g = (const float*)d_in[11];
    const float* ln2_b = (const float*)d_in[12];
    float* out = (float*)d_out;

    char* ws = (char*)d_ws;
    size_t off = 0;
    auto alloc = [&](size_t bytes) -> void* {
        void* p = ws + off;
        off += (bytes + 255) & ~(size_t)255;
        return p;
    };
    bf16* wq_b = (bf16*)alloc((size_t)D_ * D_ * 2);
    bf16* wk_b = (bf16*)alloc((size_t)D_ * D_ * 2);
    bf16* wv_b = (bf16*)alloc((size_t)D_ * D_ * 2);
    bf16* wo_b = (bf16*)alloc((size_t)D_ * D_ * 2);
    bf16* w1_b = (bf16*)alloc((size_t)DFF_ * D_ * 2);
    bf16* w2_b = (bf16*)alloc((size_t)D_ * DFF_ * 2);
    bf16* h1   = (bf16*)alloc((size_t)BS_ * D_ * 2);
    bf16* qb   = (bf16*)alloc((size_t)B_ * H_ * SP * HD_ * 2);
    bf16* kb   = (bf16*)alloc((size_t)B_ * H_ * SP * HD_ * 2);
    bf16* vtb  = (bf16*)alloc((size_t)B_ * H_ * SP * HD_ * 2);
    bf16* ctx  = (bf16*)alloc((size_t)BS_ * D_ * 2);
    float* x1  = (float*)alloc((size_t)BS_ * D_ * 4);
    bf16* h2   = (bf16*)alloc((size_t)BS_ * D_ * 2);
    bf16* act  = (bf16*)alloc((size_t)BS_ * DFF_ * 2);

    // weights -> bf16
    k_cvt_bf16<<<(D_ * D_ + 255) / 256, 256, 0, stream>>>(Wq, wq_b, D_ * D_);
    k_cvt_bf16<<<(D_ * D_ + 255) / 256, 256, 0, stream>>>(Wk, wk_b, D_ * D_);
    k_cvt_bf16<<<(D_ * D_ + 255) / 256, 256, 0, stream>>>(Wv, wv_b, D_ * D_);
    k_cvt_bf16<<<(D_ * D_ + 255) / 256, 256, 0, stream>>>(Wo, wo_b, D_ * D_);
    k_cvt_bf16<<<(DFF_ * D_ + 255) / 256, 256, 0, stream>>>(W1, w1_b, DFF_ * D_);
    k_cvt_bf16<<<(DFF_ * D_ + 255) / 256, 256, 0, stream>>>(W2, w2_b, D_ * DFF_);

    // LN1
    k_layernorm<<<BS_, 256, 0, stream>>>(hs, ln1_g, ln1_b, h1);

    // QKV projections
    dim3 gD(D_ / BN, BS_ / BM);        // (8, 125)
    k_gemm<0><<<gD, 256, 0, stream>>>(h1, wq_b, nullptr, nullptr, qb, BS_, D_, D_, 0.125f);
    k_gemm<0><<<gD, 256, 0, stream>>>(h1, wk_b, nullptr, nullptr, kb, BS_, D_, D_, 1.0f);
    k_gemm<1><<<gD, 256, 0, stream>>>(h1, wv_b, nullptr, nullptr, vtb, BS_, D_, D_, 1.0f);

    // attention
    k_attn<<<B_ * H_ * NQG, 256, 0, stream>>>(qb, kb, vtb, ctx);

    // output projection + residual
    k_gemm<2><<<gD, 256, 0, stream>>>(ctx, wo_b, nullptr, hs, x1, BS_, D_, D_, 1.0f);

    // LN2
    k_layernorm<<<BS_, 256, 0, stream>>>(x1, ln2_g, ln2_b, h2);

    // FFN
    dim3 gF(DFF_ / BN, BS_ / BM);      // (32, 125)
    k_gemm<3><<<gF, 256, 0, stream>>>(h2, w1_b, b1, nullptr, act, BS_, DFF_, D_, 1.0f);
    k_gemm<4><<<gD, 256, 0, stream>>>(act, w2_b, b2, x1, out, BS_, D_, DFF_, 1.0f);
}
